// GeneralMACE_40535901340035
// MI455X (gfx1250) — matrix operand order repaired and checked by reading, not compiled, not run on hardware
//
#include <hip/hip_runtime.h>
#include <hip/hip_bf16.h>
#include <math.h>

// ---------------------------------------------------------------------------
// Problem constants (match reference)
// ---------------------------------------------------------------------------
#define NN     10000      // nodes
#define EE     160000     // edges
#define FF     64         // features
#define NSs    10         // species
#define NBb    8          // bessel
#define LMm    16         // lm components
#define RMAXf  10.0f
#define INV_AVG (1.0f/16.0f)

typedef __attribute__((ext_vector_type(16))) _Float16 v16h;
typedef __attribute__((ext_vector_type(8)))  _Float16 v8h;
typedef __attribute__((ext_vector_type(4)))  _Float16 v4h;
typedef __attribute__((ext_vector_type(2)))  _Float16 v2h;
typedef __attribute__((ext_vector_type(8)))  float    v8f;

__constant__ int c_LOF[16] = {0,1,1,1,2,2,2,2,2,3,3,3,3,3,3,3};

__device__ __forceinline__ float silu_f(float x) {
    return x * __builtin_amdgcn_rcpf(1.0f + __expf(-x));
}

// ---------------------------------------------------------------------------
// WMMA fragment helpers (wave32, ISA 7.12.2 layouts).
// A 16x32 f16 slot order per lane (m=lane&15, g=lane>>4):
//   a[0..7]  = K = g*8   .. g*8+7     (contiguous in row-major source)
//   a[8..15] = K = 16+g*8.. 16+g*8+7  (contiguous)
// B 32x16 f16 per lane (n=lane&15, g=lane>>4): b[s] = K in A's order (g*8+s, s<8; 16+g*8+s-8, s>=8), column n.
// C 16x16 f32 per lane: c[v] = row (v + 8*(lane>>4)), col (lane&15).
// ---------------------------------------------------------------------------

// A fragment from row-major f16 source: two contiguous 16B loads.
__device__ __forceinline__ v16h ldA_f16row(const _Float16* src, int ld) {
    int lane = threadIdx.x & 31;
    int m = lane & 15, g = lane >> 4;
    const v8h* p0 = (const v8h*)(src + m * ld + g * 8);
    const v8h* p1 = (const v8h*)(src + m * ld + 16 + g * 8);
    v8h lo = *p0, hi = *p1;
    return __builtin_shufflevector(lo, hi, 0,1,2,3,4,5,6,7,8,9,10,11,12,13,14,15);
}

// A or B fragment from pre-swizzled LDS (frag-linear [lane][16]): 32B load.
__device__ __forceinline__ v16h ld_swz(const _Float16* base) {
    int lane = threadIdx.x & 31;
    const v8h* p = (const v8h*)(base + lane * 16);
    v8h lo = p[0], hi = p[1];
    return __builtin_shufflevector(lo, hi, 0,1,2,3,4,5,6,7,8,9,10,11,12,13,14,15);
}

// Pack 8 scaled f32 -> 8 f16 with v_cvt_pk_rtz_f16_f32 (2 per instruction).
__device__ __forceinline__ v8h pk8(float4 a, float4 b, float s) {
    v2h p0 = __builtin_bit_cast(v2h, __builtin_amdgcn_cvt_pkrtz(a.x * s, a.y * s));
    v2h p1 = __builtin_bit_cast(v2h, __builtin_amdgcn_cvt_pkrtz(a.z * s, a.w * s));
    v2h p2 = __builtin_bit_cast(v2h, __builtin_amdgcn_cvt_pkrtz(b.x * s, b.y * s));
    v2h p3 = __builtin_bit_cast(v2h, __builtin_amdgcn_cvt_pkrtz(b.z * s, b.w * s));
    v4h q0 = __builtin_shufflevector(p0, p1, 0,1,2,3);
    v4h q1 = __builtin_shufflevector(p2, p3, 0,1,2,3);
    return __builtin_shufflevector(q0, q1, 0,1,2,3,4,5,6,7);
}

// A fragment from row-contiguous f32 source (element (m,k) = src[m*rs+k]),
// scaled + packed-converted: four 16B loads, eight pk-cvt per lane.
__device__ __forceinline__ v16h ldA_f32row(const float* src, int rs, float scale) {
    int lane = threadIdx.x & 31;
    int m = lane & 15, g = lane >> 4;
    const float4* q0 = (const float4*)(src + m * rs + g * 8);
    const float4* q1 = (const float4*)(src + m * rs + 16 + g * 8);
    v8h lo = pk8(q0[0], q0[1], scale);
    v8h hi = pk8(q1[0], q1[1], scale);
    return __builtin_shufflevector(lo, hi, 0,1,2,3,4,5,6,7,8,9,10,11,12,13,14,15);
}

// Swizzle index for B-fragment storage: element (k,n) of a K x (ntiles*16)
// matrix -> frag-linear offset. Frag = (k>>5)*ntiles + (n>>4), 512 f16 each.
__device__ __forceinline__ int bswz_idx(int k, int n, int ntiles) {
    int kc = k >> 5, nt = n >> 4;
    int lane = (((k >> 3) & 1) << 4) | (n & 15);
    int slot = (((k >> 4) & 1) << 3) | (k & 7);
    return ((kc * ntiles + nt) * 32 + lane) * 16 + slot;
}

// Swizzle index for A-fragment storage of a TRANSPOSED source:
// element (kf, gcol) of Wsc (K=64 rows f, 64 cols g) -> A-frag-linear offset.
__device__ __forceinline__ int aswz_idx(int kf, int gcol) {
    int ks = kf >> 5, mt = gcol >> 4;
    int kp = kf & 31;
    int gA = (kp >> 3) & 1;
    int hi = (kp >> 4) & 1;
    int slot = hi * 8 + (kp - hi * 16 - gA * 8);
    int lane = gA * 16 + (gcol & 15);
    return ((ks * 4 + mt) * 32 + lane) * 16 + slot;
}

__device__ __forceinline__ void stC_f16(_Float16* base, int rs, v8f c) {
    int lane = threadIdx.x & 31;
    int n = lane & 15, g = lane >> 4;
#pragma unroll
    for (int v = 0; v < 8; ++v)
        base[(v + 8 * g) * rs + n] = (_Float16)c[v];
}

__device__ __forceinline__ void stC_f16_silu(_Float16* base, int rs, v8f c) {
    int lane = threadIdx.x & 31;
    int n = lane & 15, g = lane >> 4;
#pragma unroll
    for (int v = 0; v < 8; ++v)
        base[(v + 8 * g) * rs + n] = (_Float16)silu_f(c[v]);
}

#define WMMA_F16(a, b, c) \
    __builtin_amdgcn_wmma_f32_16x16x32_f16(false, (a), false, (b), (short)0, (c), false, false)

// ---------------------------------------------------------------------------
// Kernel 1: per-edge geometry -> Y [E,16] f32, bessel padded [E,32] f16
// ---------------------------------------------------------------------------
__global__ __launch_bounds__(256)
void edge_geom_kernel(const float* __restrict__ pos,
                      const float* __restrict__ shifts,
                      const int* __restrict__ senders,
                      const int* __restrict__ receivers,
                      float* __restrict__ Yout,
                      _Float16* __restrict__ bessp) {
    int e = blockIdx.x * 256 + threadIdx.x;
    if (e >= EE) return;
    int s = senders[e], r = receivers[e];
    float vx = pos[r*3+0] - pos[s*3+0] + shifts[e*3+0];
    float vy = pos[r*3+1] - pos[s*3+1] + shifts[e*3+1];
    float vz = pos[r*3+2] - pos[s*3+2] + shifts[e*3+2];
    float rr = sqrtf(vx*vx + vy*vy + vz*vz);
    float inv = __builtin_amdgcn_rcpf(rr + 1e-9f);
    float x = vx * inv, y = vy * inv, z = vz * inv;

    const float s3 = 1.73205081f, s5 = 2.23606798f, s7 = 2.64575131f;
    const float s15 = 3.87298335f, s42 = 6.48074069f, s70 = 8.36660027f, s105 = 10.24695077f;
    float x2 = x*x, y2 = y*y, z2 = z*z;
    float4* Yp = (float4*)(Yout + (size_t)e * 16);
    Yp[0] = make_float4(1.0f, s3 * x, s3 * y, s3 * z);
    Yp[1] = make_float4(s15 * x * y, s15 * y * z,
                        0.5f * s5 * (3.0f * z2 - 1.0f), s15 * x * z);
    Yp[2] = make_float4(0.5f * s15 * (x2 - y2),
                        0.25f * s70 * y * (3.0f * x2 - y2),
                        s105 * x * y * z,
                        0.25f * s42 * y * (5.0f * z2 - 1.0f));
    Yp[3] = make_float4(0.5f * s7 * z * (5.0f * z2 - 3.0f),
                        0.25f * s42 * x * (5.0f * z2 - 1.0f),
                        0.5f * s105 * z * (x2 - y2),
                        0.25f * s70 * x * (x2 - 3.0f * y2));

    // bessel + polynomial cutoff envelope (p=5)
    float xr = rr / RMAXf;
    float xr2 = xr * xr, xr5 = xr2 * xr2 * xr;
    float env = 1.0f - 21.0f * xr5 + 35.0f * xr5 * xr - 15.0f * xr5 * xr2;
    if (xr >= 1.0f) env = 0.0f;
    float amp = 0.44721360f /* sqrt(2/10) */ * inv * env;
    const float PI = 3.14159265358979f;
    v8h b0;
#pragma unroll
    for (int n = 1; n <= NBb; ++n)
        b0[n-1] = (_Float16)(amp * __sinf((float)n * PI * rr / RMAXf));
    v8h zz = {};
    v8h* bp = (v8h*)(bessp + (size_t)e * 32);
    bp[0] = b0; bp[1] = zz; bp[2] = zz; bp[3] = zz;
}

// ---------------------------------------------------------------------------
// Kernel 2: species argmax from one-hot attrs
// ---------------------------------------------------------------------------
__global__ __launch_bounds__(256)
void spec_kernel(const float* __restrict__ attrs, int* __restrict__ spec) {
    int n = blockIdx.x * 256 + threadIdx.x;
    if (n >= NN) return;
    float best = -1.0f; int bi = 0;
#pragma unroll
    for (int s = 0; s < NSs; ++s) {
        float v = attrs[n * NSs + s];
        if (v > best) { best = v; bi = s; }
    }
    spec[n] = bi;
}

// ---------------------------------------------------------------------------
// Kernel 3: embed.  nf layout is k-major: [N][16 k][64 f].
// ---------------------------------------------------------------------------
__global__ __launch_bounds__(256)
void embed_kernel(const float* __restrict__ attrs,
                  const float* __restrict__ wembed,
                  float* __restrict__ nf) {
    int idx = blockIdx.x * 256 + threadIdx.x;
    if (idx >= NN * FF) return;
    int n = idx >> 6, f = idx & 63;
    float acc = 0.0f;
#pragma unroll
    for (int s = 0; s < NSs; ++s)
        acc += attrs[n * NSs + s] * wembed[s * FF + f];
    float* p = nf + (size_t)n * 1024;
    p[f] = acc;
#pragma unroll
    for (int k = 1; k < 16; ++k) p[k * 64 + f] = 0.0f;
}

// ---------------------------------------------------------------------------
// Kernel 4: radial MLP via WMMA.  Block = 128 thr (4 waves), wave = 16 edges.
// A-fragments hoisted out of the output-tile loops.
// ---------------------------------------------------------------------------
__global__ __launch_bounds__(128)
void radial_kernel(const _Float16* __restrict__ bessp,
                   const float* __restrict__ Wr1,   // [8][64]
                   const float* __restrict__ Wr2,   // [64][64]
                   const float* __restrict__ Wr3,   // [64][256]
                   _Float16* __restrict__ Rout) {
    __shared__ alignas(16) _Float16 sW1[32 * 64];    //  4 frags
    __shared__ alignas(16) _Float16 sW2[64 * 64];    //  8 frags
    __shared__ alignas(16) _Float16 sW3[64 * 256];   // 32 frags
    __shared__ alignas(16) _Float16 sStg[4][2][16 * 64];

    int t = threadIdx.x;
    for (int i = t; i < 32 * 64; i += 128) {
        int k = i >> 6, n = i & 63;
        sW1[bswz_idx(k, n, 4)] = (k < NBb) ? (_Float16)Wr1[k * 64 + n] : (_Float16)0.0f;
    }
    for (int i = t; i < 64 * 64; i += 128) {
        int k = i >> 6, n = i & 63;
        sW2[bswz_idx(k, n, 4)] = (_Float16)Wr2[i];
    }
    for (int i = t; i < 64 * 256; i += 128) {
        int k = i >> 8, n = i & 255;
        sW3[bswz_idx(k, n, 16)] = (_Float16)Wr3[i];
    }
    __syncthreads();

    int wid = t >> 5;
    int tile = blockIdx.x * 4 + wid;            // uniform per wave
    if (tile * 16 >= EE) return;

    _Float16* st0 = &sStg[wid][0][0];
    _Float16* st1 = &sStg[wid][1][0];

    // layer 1: [16x32(8)] x [32x64]
    v16h a0 = ldA_f16row(bessp + (size_t)tile * 16 * 32, 32);
#pragma unroll
    for (int nt = 0; nt < 4; ++nt) {
        v8f acc = {};
        acc = WMMA_F16(a0, ld_swz(sW1 + nt * 512), acc);
        stC_f16_silu(st0 + nt * 16, 64, acc);
    }
    // layer 2: [16x64] x [64x64]   (A frags loaded once)
    v16h a20 = ldA_f16row(st0, 64);
    v16h a21 = ldA_f16row(st0 + 32, 64);
#pragma unroll
    for (int nt = 0; nt < 4; ++nt) {
        v8f acc = {};
        acc = WMMA_F16(a20, ld_swz(sW2 + nt * 512), acc);
        acc = WMMA_F16(a21, ld_swz(sW2 + (4 + nt) * 512), acc);
        stC_f16_silu(st1 + nt * 16, 64, acc);
    }
    // layer 3: [16x64] x [64x256]  (A frags loaded once)
    v16h a30 = ldA_f16row(st1, 64);
    v16h a31 = ldA_f16row(st1 + 32, 64);
    _Float16* out = Rout + (size_t)tile * 16 * 256;
#pragma unroll
    for (int nt = 0; nt < 16; ++nt) {
        v8f acc = {};
        acc = WMMA_F16(a30, ld_swz(sW3 + nt * 512), acc);
        acc = WMMA_F16(a31, ld_swz(sW3 + (16 + nt) * 512), acc);
        stC_f16(out + nt * 16, 256, acc);
    }
}

// ---------------------------------------------------------------------------
// Kernel 5: h = nf[:,0,:] @ W_up  via WMMA -> h [N][64] f16
// ---------------------------------------------------------------------------
__global__ __launch_bounds__(128)
void h_kernel(const float* __restrict__ nf,
              const float* __restrict__ Wup,    // [64][64]
              _Float16* __restrict__ hOut) {
    __shared__ alignas(16) _Float16 sW[64 * 64];
    int t = threadIdx.x;
    for (int i = t; i < 4096; i += 128) {
        int k = i >> 6, n = i & 63;
        sW[bswz_idx(k, n, 4)] = (_Float16)Wup[i];
    }
    __syncthreads();

    int tile = blockIdx.x * 4 + (t >> 5);
    if (tile * 16 >= NN) return;
    int n0 = tile * 16;

    v16h a0 = ldA_f32row(nf + (size_t)n0 * 1024, 1024, 1.0f);
    v16h a1 = ldA_f32row(nf + (size_t)n0 * 1024 + 32, 1024, 1.0f);
#pragma unroll
    for (int nt = 0; nt < 4; ++nt) {
        v8f acc = {};
        acc = WMMA_F16(a0, ld_swz(sW + nt * 512), acc);
        acc = WMMA_F16(a1, ld_swz(sW + (4 + nt) * 512), acc);
        stC_f16(hOut + (size_t)n0 * 64 + nt * 16, 64, acc);
    }
}

// ---------------------------------------------------------------------------
// Kernel 6: message + scatter.  Thread = (edge, f).  16 f32 atomics each.
// A layout k-major: A[recv][k][f] += h[send,f] * R[e,f,L_OF[k]] * Y[e,k]
// ---------------------------------------------------------------------------
__global__ __launch_bounds__(256)
void scatter_kernel(const _Float16* __restrict__ R,
                    const _Float16* __restrict__ h,
                    const float* __restrict__ Y,
                    const int* __restrict__ senders,
                    const int* __restrict__ receivers,
                    float* __restrict__ A) {
    int idx = blockIdx.x * 256 + threadIdx.x;
    if (idx >= EE * FF) return;
    int e = idx >> 6, f = idx & 63;
    int s = senders[e], r = receivers[e];
    __builtin_prefetch(h + (size_t)s * 64 + f, 0, 0);   // global_prefetch_b8
    v4h rv = *(const v4h*)(R + (size_t)e * 256 + f * 4);
    float Rl[4];
#pragma unroll
    for (int l = 0; l < 4; ++l) Rl[l] = (float)rv[l];
    float hs = (float)h[(size_t)s * 64 + f];
    const float4* Yv = (const float4*)(Y + (size_t)e * 16);
    float4 y0 = Yv[0], y1 = Yv[1], y2 = Yv[2], y3 = Yv[3];
    float yk[16] = { y0.x,y0.y,y0.z,y0.w, y1.x,y1.y,y1.z,y1.w,
                     y2.x,y2.y,y2.z,y2.w, y3.x,y3.y,y3.z,y3.w };
    float* ap = A + (size_t)r * 1024 + f;
#pragma unroll
    for (int k = 0; k < 16; ++k) {
        float m = hs * Rl[c_LOF[k]] * yk[k];
        unsafeAtomicAdd(ap + k * 64, m);
    }
}

// ---------------------------------------------------------------------------
// Kernel 7: channel mix via WMMA: Amix[n,k,g] = (A[n,k,f]/AVG) @ Wmix[L_OF[k]]
// Wave = 16-node tile; A frags per k-slice loaded once (hoisted).
// ---------------------------------------------------------------------------
__global__ __launch_bounds__(128)
void mix_kernel(const float* __restrict__ A,
                const float* __restrict__ Wmix,  // [4][64][64]
                float* __restrict__ Amix) {
    __shared__ alignas(16) _Float16 sW[4 * 64 * 64]; // 4 l x 8 frags
    int t = threadIdx.x;
    for (int i = t; i < 4 * 4096; i += 128) {
        int l = i >> 12, rem = i & 4095;
        int k = rem >> 6, n = rem & 63;
        sW[l * 4096 + bswz_idx(k, n, 4)] = (_Float16)Wmix[i];
    }
    __syncthreads();

    int tile = blockIdx.x * 4 + (t >> 5);
    if (tile * 16 >= NN) return;
    int n0 = tile * 16;
    int lane = t & 31, nl = lane & 15, g = lane >> 4;

#pragma unroll
    for (int k = 0; k < 16; ++k) {
        const _Float16* W = sW + c_LOF[k] * 4096;
        v16h aA = ldA_f32row(A + (size_t)n0 * 1024 + k * 64, 1024, INV_AVG);
        v16h aB = ldA_f32row(A + (size_t)n0 * 1024 + k * 64 + 32, 1024, INV_AVG);
#pragma unroll
        for (int nt = 0; nt < 4; ++nt) {
            v8f acc = {};
            acc = WMMA_F16(aA, ld_swz(W + nt * 512), acc);
            acc = WMMA_F16(aB, ld_swz(W + (4 + nt) * 512), acc);
            float* base = Amix + (size_t)n0 * 1024 + k * 64 + nt * 16;
#pragma unroll
            for (int v = 0; v < 8; ++v)
                base[(size_t)(v + 8 * g) * 1024 + nl] = acc[v];
        }
    }
}

// ---------------------------------------------------------------------------
// Kernel 8: per-node self-connection (WMMA, species weight pre-swizzled to
// A-fragment order in LDS) + polynomial gate.  Wave = 1 node.
// ---------------------------------------------------------------------------
__global__ __launch_bounds__(128)
void scpoly_kernel(const float* __restrict__ nf,
                   const float* __restrict__ Amix,
                   const float* __restrict__ Wsc,    // [NS][64][64]
                   const float* __restrict__ wprod,  // [3][64]
                   const int* __restrict__ spec,
                   float* __restrict__ nfOut,
                   int lastFlag) {
    __shared__ alignas(16) _Float16 sWsc[4][64 * 64]; // A-swizzled, 8 frags
    __shared__ alignas(16) _Float16 sX[4][64 * 16];   // B-swizzled, 2 frags
    int t = threadIdx.x, wid = t >> 5, lane = t & 31;
    int node = blockIdx.x * 4 + wid;                // uniform per wave
    if (node >= NN) return;

    const float* wp = Wsc + (size_t)spec[node] * 4096;
    for (int i = lane; i < 4096; i += 32) {
        int f = i >> 6, gcol = i & 63;
        sWsc[wid][aswz_idx(f, gcol)] = (_Float16)wp[i];
    }
    const float* nfp = nf + (size_t)node * 1024;   // [k][f] k-major
    for (int i = lane; i < 1024; i += 32) {
        int k = i >> 6, f = i & 63;
        sX[wid][bswz_idx(f, k, 1)] = (_Float16)nfp[i];
    }
    // same-wave LDS ops are in-order (DScnt); no cross-wave sharing here.

    const _Float16* W = sWsc[wid];
    const float* am = Amix + (size_t)node * 1024;   // [k][g]
    float* outp = nfOut + (size_t)node * 1024;
    int nl = lane & 15, gg = lane >> 4;

    v16h b0 = ld_swz(sX[wid]);
    v16h b1 = ld_swz(sX[wid] + 512);

#pragma unroll
    for (int mt = 0; mt < 4; ++mt) {
        v8f acc = {};
        acc = WMMA_F16(ld_swz(W + mt * 512), b0, acc);
        acc = WMMA_F16(ld_swz(W + (4 + mt) * 512), b1, acc);
        // c[v]: row g = mt*16 + v + 8*gg (v consecutive), col k = nl
        float4 o[2];
#pragma unroll
        for (int v = 0; v < 8; ++v) {
            int gidx = mt * 16 + v + 8 * gg;
            float s = am[gidx];                       // k=0 row
            float poly = wprod[gidx] + wprod[64 + gidx] * s
                       + wprod[128 + gidx] * s * s;
            float val = am[nl * 64 + gidx] * poly + acc[v];
            if (lastFlag && nl != 0) val = 0.0f;
            ((float*)o)[v] = val;
        }
        float4* op = (float4*)(outp + nl * 64 + mt * 16 + 8 * gg);
        op[0] = o[0]; op[1] = o[1];
    }
}

// ---------------------------------------------------------------------------
// Kernel 9/10: readouts into d_out [N][2]
// ---------------------------------------------------------------------------
__global__ __launch_bounds__(256)
void read0_kernel(const float* __restrict__ nf,
                  const float* __restrict__ wread,  // [64]
                  float* __restrict__ out) {
    int n = blockIdx.x * 256 + threadIdx.x;
    if (n >= NN) return;
    const float* p = nf + (size_t)n * 1024;         // k=0 row contiguous
    float acc = 0.0f;
#pragma unroll
    for (int f = 0; f < FF; ++f)
        acc += p[f] * wread[f];
    out[n * 2 + 0] = acc;
}

__global__ __launch_bounds__(256)
void read1_kernel(const float* __restrict__ nf,
                  const float* __restrict__ w1,   // [64][16]
                  const float* __restrict__ w2,   // [16]
                  float* __restrict__ out) {
    int n = blockIdx.x * 256 + threadIdx.x;
    if (n >= NN) return;
    const float* p = nf + (size_t)n * 1024;         // k=0 row contiguous
    float x[FF];
#pragma unroll
    for (int f = 0; f < FF; ++f) x[f] = p[f];
    float acc = 0.0f;
#pragma unroll
    for (int j = 0; j < 16; ++j) {
        float z = 0.0f;
#pragma unroll
        for (int f = 0; f < FF; ++f) z += x[f] * w1[f * 16 + j];
        acc += silu_f(z) * w2[j];
    }
    out[n * 2 + 1] = acc;
}

// ---------------------------------------------------------------------------
// Host orchestration
// ---------------------------------------------------------------------------
extern "C" void kernel_launch(void* const* d_in, const int* in_sizes, int n_in,
                              void* d_out, int out_size, void* d_ws, size_t ws_size,
                              hipStream_t stream) {
    (void)in_sizes; (void)n_in; (void)out_size; (void)ws_size;

    const float* pos    = (const float*)d_in[0];
    const float* attrs  = (const float*)d_in[1];
    const float* shifts = (const float*)d_in[2];
    const int*   send   = (const int*)d_in[3];
    const int*   recv   = (const int*)d_in[4];
    const float* wembed = (const float*)d_in[5];
    const float* wread0 = (const float*)d_in[20];
    const float* wmlp1  = (const float*)d_in[21];
    const float* wmlp2  = (const float*)d_in[22];
    float* out = (float*)d_out;

    char* ws = (char*)d_ws;
    size_t off = 0;
    auto carve = [&](size_t bytes) -> char* {
        char* p = ws + off;
        off = (off + bytes + 255) & ~(size_t)255;
        return p;
    };
    float*    Y     = (float*)   carve((size_t)EE * 16 * 4);
    _Float16* bessp = (_Float16*)carve((size_t)EE * 32 * 2);
    _Float16* R     = (_Float16*)carve((size_t)EE * 256 * 2);
    _Float16* hbuf  = (_Float16*)carve((size_t)NN * 64 * 2);
    float*    A     = (float*)   carve((size_t)NN * 1024 * 4);
    float*    Amix  = (float*)   carve((size_t)NN * 1024 * 4);
    float*    nfA   = (float*)   carve((size_t)NN * 1024 * 4);
    float*    nfB   = (float*)   carve((size_t)NN * 1024 * 4);
    int*      spec  = (int*)     carve((size_t)NN * 4);

    // --- shared precompute ---
    edge_geom_kernel<<<(EE + 255) / 256, 256, 0, stream>>>(pos, shifts, send, recv, Y, bessp);
    spec_kernel<<<(NN + 255) / 256, 256, 0, stream>>>(attrs, spec);
    embed_kernel<<<(NN * FF + 255) / 256, 256, 0, stream>>>(attrs, wembed, nfA);

    const int radialBlocks   = (EE / 16 + 3) / 4;        // 2500
    const int nodeTileBlocks = ((NN + 15) / 16 + 3) / 4; // 157
    const int scatterBlocks  = (EE * FF) / 256;          // 40000
    const int nodeWaveBlocks = (NN + 3) / 4;             // 2500

    float* nf_in = nfA;
    float* nf_out = nfB;
    for (int layer = 0; layer < 2; ++layer) {
        int base = 6 + layer * 7;
        const float* Wup  = (const float*)d_in[base + 0];
        const float* Wr1  = (const float*)d_in[base + 1];
        const float* Wr2  = (const float*)d_in[base + 2];
        const float* Wr3  = (const float*)d_in[base + 3];
        const float* Wmix = (const float*)d_in[base + 4];
        const float* Wsc  = (const float*)d_in[base + 5];
        const float* Wpr  = (const float*)d_in[base + 6];
        int last = (layer == 1) ? 1 : 0;

        radial_kernel<<<radialBlocks, 128, 0, stream>>>(bessp, Wr1, Wr2, Wr3, R);
        h_kernel<<<nodeTileBlocks, 128, 0, stream>>>(nf_in, Wup, hbuf);
        (void)hipMemsetAsync(A, 0, (size_t)NN * 1024 * 4, stream);
        scatter_kernel<<<scatterBlocks, 256, 0, stream>>>(R, hbuf, Y, send, recv, A);
        mix_kernel<<<nodeTileBlocks, 128, 0, stream>>>(A, Wmix, Amix);
        scpoly_kernel<<<nodeWaveBlocks, 128, 0, stream>>>(nf_in, Amix, Wsc, Wpr, spec, nf_out, last);

        if (layer == 0)
            read0_kernel<<<(NN + 255) / 256, 256, 0, stream>>>(nf_out, wread0, out);

        float* tmp = nf_in; nf_in = nf_out; nf_out = tmp;
    }
    // after loop nf_in points at final features (nfA)
    read1_kernel<<<(NN + 255) / 256, 256, 0, stream>>>(nf_in, wmlp1, wmlp2, out);
}
